// GTTensorFieldNetwork_36043365548222
// MI455X (gfx1250) — matrix operand order repaired and checked by reading, not compiled, not run on hardware
//
#include <hip/hip_runtime.h>
#include <math.h>

// ---------------------------------------------------------------------------
// GT Tensor Field Network forward for MI455X (gfx1250, wave32, WMMA).
//
// FLOPs dominated by radial-MLP GEMMs (~46 GFLOP): mapped to
// v_wmma_f32_16x16x32_f16 (f16 in, f32 acc). The O(N^2*256) "rad" tensor
// never leaves LDS: one workgroup per (batch, node i) streams 16-j tiles
// (h1 -> rad -> analytic-CG contraction). rbf is f16 with K padded 16->32 so
// one K=32 WMMA covers both MLP layers. h1 for all 256 j-rows is computed
// once per interaction (32 WMMAs over all 8 waves). The tile loop is
// software-pipelined to 2 barriers/tile, and s_rad uses a 272-float row
// stride so the contraction's ds_loads are bank-conflict free (64 banks).
// All global intermediates (~20 MB) fit in the 192 MB L2 -> matrix-bound.
// ---------------------------------------------------------------------------

#define N_PTS 256
#define BATCH 4
#define NH    16      // hidden channels H
#define KPAD  32      // rbf padded K (16 real + 16 zero)
#define RADLD 272     // s_rad row stride (floats): 256 + 16 -> conflict-free
#define EPSV  1e-8f

typedef __attribute__((ext_vector_type(16))) _Float16 v16h;
typedef __attribute__((ext_vector_type(8)))  float    v8f;

// ---- WMMA fragment helpers (contiguous K order, same for A and B, wave32) ----------

// A-matrix 16x32 f16: lane l -> M = l&15, half = l>>4;
// elems 0..15 = K[half*16+e].
__device__ __forceinline__ v16h frag_a16x32(const _Float16* base, int ldm, int lane) {
    int m = lane & 15, hf = lane >> 4;
    const _Float16* r = base + m * ldm + hf * 16;
    union { int4 q[2]; v16h v; } u;
    u.q[0] = *(const int4*)(r);        // K = hf*16 + 0..7
    u.q[1] = *(const int4*)(r + 8);    // K = hf*16 + 8..15
    return u.v;
}

// B-matrix 32x16 f16 staged in LDS in exact fragment order:
// per tile: [lane][elem], elem e -> K = (lane>>4)*16 + e, N = lane&15.
__device__ __forceinline__ v16h frag_b_staged(const _Float16* p) {
    union { int4 q[2]; v16h v; } u;
    u.q[0] = ((const int4*)p)[0];
    u.q[1] = ((const int4*)p)[1];
    return u.v;
}

__device__ __forceinline__ float fast_rcp(float x) { return __builtin_amdgcn_rcpf(x); }

struct LayerParams {
    const float* W1[5];   // (16,32)
    const float* b1[5];   // (32)
    const float* W2[5];   // (32, c_in*16)
    const float* b2[5];   // (c_in*16)
    const float* ln_g[2]; // s, v  (16)
    const float* ln_b[2]; // s, v  (16)
};

// ---------------------------------------------------------------------------
// Geometry: rbf (f16, K-padded), r_hat, initial node features.
// grid = B*N blocks, 256 threads; block = (b, i), thread = j.
// ---------------------------------------------------------------------------
__global__ void __launch_bounds__(256)
geom_kernel(const float* __restrict__ pos,
            _Float16* __restrict__ rbf, float* __restrict__ rhat,
            float* __restrict__ s0, float* __restrict__ v0) {
    int blk = blockIdx.x;
    int b = blk / N_PTS, i = blk % N_PTS;
    int j = threadIdx.x;
    const float* pi = pos + (size_t)(b * N_PTS + i) * 3;
    const float* pj = pos + (size_t)(b * N_PTS + j) * 3;
    float dx = pi[0] - pj[0], dy = pi[1] - pj[1], dz = pi[2] - pj[2];
    float dist = sqrtf(dx * dx + dy * dy + dz * dz);
    float inv = fast_rcp(fmaxf(dist, EPSV));
    size_t pidx = (size_t)(b * N_PTS + i) * N_PTS + j;
    rhat[pidx * 3 + 0] = dx * inv;
    rhat[pidx * 3 + 1] = dy * inv;
    rhat[pidx * 3 + 2] = dz * inv;
    _Float16* rr = rbf + pidx * KPAD;
    const float step = 5.0f / 15.0f;                 // linspace(0,5,16)
    const float invw = 1.0f / 0.09765625f;           // (5/16)^2
#pragma unroll
    for (int k = 0; k < 16; ++k) {
        float d = dist - step * (float)k;
        rr[k] = (_Float16)__expf(-d * d * invw);
    }
#pragma unroll
    for (int k = 16; k < 32; ++k) rr[k] = (_Float16)0.0f;   // K padding
    if (j == 0) {
        float pn = sqrtf(pi[0] * pi[0] + pi[1] * pi[1] + pi[2] * pi[2]);
        s0[b * N_PTS + i] = pn;
        float ip = fast_rcp(fmaxf(pn, EPSV));
        v0[(b * N_PTS + i) * 3 + 0] = pi[0] * ip;
        v0[(b * N_PTS + i) * 3 + 1] = pi[1] * ip;
        v0[(b * N_PTS + i) * 3 + 2] = pi[2] * ip;
    }
}

// ---------------------------------------------------------------------------
// One message-passing layer, specialized on CIN (1 for layer 0, 16 after).
// grid = (N, B), 256 threads (8 waves); block = (node i, batch b).
// ---------------------------------------------------------------------------
template <int CIN>
__global__ void __launch_bounds__(256)
layer_kernel(const _Float16* __restrict__ rbf, const float* __restrict__ rhat,
             const float* __restrict__ s_in, const float* __restrict__ v_in,
             float* __restrict__ s_out, float* __restrict__ v_out,
             LayerParams P) {
    constexpr int NCOL  = CIN * NH;     // 16 or 256
    constexpr int NTILE = NCOL >> 4;    // 1 or 16

    __shared__ __align__(16) _Float16 s_rbf[N_PTS][KPAD];   // 16 KB
    __shared__ __align__(16) _Float16 s_W1f[2 * 512];       //  2 KB (frag order)
    __shared__ __align__(16) _Float16 s_W2f[16 * 512];      // 16 KB (frag order)
    __shared__ __align__(16) _Float16 s_h1[N_PTS][KPAD];    // 16 KB (full h1)
    __shared__ float s_b1[32];
    __shared__ float s_b2[256];
    __shared__ float s_rad[16][RADLD];                      // 17 KB (padded)
    __shared__ float s_G[16][16][3];                        //  3 KB
    __shared__ float s_rh[N_PTS][3];                        //  3 KB
    __shared__ float s_part[16][16][3];                     //  3 KB
    __shared__ float s_msg_s[16];
    __shared__ float s_msg_v[16][3];
    __shared__ float s_norm[2][16];
    __shared__ float s_mu[2], s_isig[2];

    const int t = threadIdx.x, lane = t & 31, wid = t >> 5;
    const int b = blockIdx.y, i = blockIdx.x;
    const size_t pbase = (size_t)(b * N_PTS + i) * N_PTS;

    // stage this node's rbf rows + r_hat row
    {
        const int4* src = (const int4*)(rbf + (pbase + t) * KPAD);
        int4* dst = (int4*)(&s_rbf[t][0]);
        dst[0] = src[0]; dst[1] = src[1]; dst[2] = src[2]; dst[3] = src[3];
        const float* rs = rhat + (pbase + t) * 3;
        s_rh[t][0] = rs[0]; s_rh[t][1] = rs[1]; s_rh[t][2] = rs[2];
    }
    if (t < 16) {
        s_msg_s[t] = 0.0f;
        s_msg_v[t][0] = 0.0f; s_msg_v[t][1] = 0.0f; s_msg_v[t][2] = 0.0f;
    }
    __syncthreads();

    for (int it = 0; it < 5; ++it) {   // sss, svv, vsv, vvs, vvv
        const float* W1 = P.W1[it]; const float* b1 = P.b1[it];
        const float* W2 = P.W2[it]; const float* b2 = P.b2[it];
        // stage W1 (K padded to 32) in B-fragment order
        for (int u = t; u < 1024; u += 256) {
            int tau = u >> 9, l = (u >> 4) & 31, e = u & 15;
            int k = ((l >> 4) << 4) + e, n = (tau << 4) + (l & 15);
            s_W1f[u] = (_Float16)((k < 16) ? W1[k * 32 + n] : 0.0f);
        }
        if (t < 32) s_b1[t] = b1[t];
        // stage W2 in B-fragment order
        for (int u = t; u < NTILE * 512; u += 256) {
            int tau = u >> 9, l = (u >> 4) & 31, e = u & 15;
            int k = ((l >> 4) << 4) + e, n = (tau << 4) + (l & 15);
            s_W2f[u] = (_Float16)W2[k * NCOL + n];
        }
        if (t < NCOL) s_b2[t] = b2[t];
        __syncthreads();

        // speculative prefetch of the next interaction's W2 into GL2
        // (global_prefetch_b8); hidden behind the work below.
        if (it < 4 && t < 64) __builtin_prefetch(P.W2[it + 1] + t * 128, 0, 1);

        // ---- full h1 = silu(rbf @ W1 + b1): 32 WMMAs over 8 waves ----
#pragma unroll
        for (int q = 0; q < 4; ++q) {
            int idx = wid * 4 + q;            // 0..31
            int m = idx >> 1, nt = idx & 1;   // M-tile (16 j), N-tile (16 cols)
            v16h a  = frag_a16x32(&s_rbf[m * 16][0], KPAD, lane);
            v16h bf = frag_b_staged(&s_W1f[nt * 512 + lane * 16]);
            v8f acc = {};
            acc = __builtin_amdgcn_wmma_f32_16x16x32_f16(
                    false, a, false, bf, (short)0, acc, false, false);
            int n = nt * 16 + (lane & 15);
            int mb = (lane >> 4) << 3;        // D: M = k + 8*half
#pragma unroll
            for (int k = 0; k < 8; ++k) {
                float x = acc[k] + s_b1[n];
                float h = x * fast_rcp(1.0f + __expf(-x));   // silu via v_rcp
                s_h1[m * 16 + k + mb][n] = (_Float16)h;
            }
        }
        __syncthreads();

        const int dout = (it == 0 || it == 3) ? 1 : 3;

        // ---- software-pipelined j-tile loop: 2 barriers per tile ----
        for (int jt = 0; jt < 16; ++jt) {
            // phase 1: G(jt) + rad-WMMA(jt) + tree-sum of s_part(jt-1)
            {
                int jl = t >> 4, c = t & 15;
                int j = jt * 16 + jl;
                float g0 = 0.0f, g1 = 0.0f, g2 = 0.0f;
                if (c < CIN && j != i) {            // mask = ~eye
                    float r0 = s_rh[j][0], r1 = s_rh[j][1], r2 = s_rh[j][2];
                    if (it == 0) {
                        g0 = s_in[(b * N_PTS + j) * CIN + c];
                    } else if (it == 1) {
                        float sv = s_in[(b * N_PTS + j) * CIN + c];
                        g0 = sv * r0; g1 = sv * r1; g2 = sv * r2;
                    } else {
                        const float* vv = v_in + ((size_t)(b * N_PTS + j) * CIN + c) * 3;
                        float vx = vv[0], vy = vv[1], vz = vv[2];
                        if (it == 2) {              // identity CG
                            g0 = vx; g1 = vy; g2 = vz;
                        } else if (it == 3) {       // dot / sqrt(3)
                            g0 = (vx * r0 + vy * r1 + vz * r2) * 0.57735026919f;
                        } else {                    // cross / sqrt(2)
                            g0 = (vy * r2 - vz * r1) * 0.70710678118f;
                            g1 = (vz * r0 - vx * r2) * 0.70710678118f;
                            g2 = (vx * r1 - vy * r0) * 0.70710678118f;
                        }
                    }
                }
                s_G[jl][c][0] = g0; s_G[jl][c][1] = g1; s_G[jl][c][2] = g2;
            }
#pragma unroll
            for (int tt = 0; tt < 2; ++tt) {        // rad = h1 @ W2 + b2
                int tau = wid * 2 + tt;
                if (tau < NTILE) {
                    v16h a  = frag_a16x32(&s_h1[jt * 16][0], KPAD, lane);
                    v16h bf = frag_b_staged(&s_W2f[tau * 512 + lane * 16]);
                    v8f acc = {};
                    acc = __builtin_amdgcn_wmma_f32_16x16x32_f16(
                            false, a, false, bf, (short)0, acc, false, false);
                    int n = tau * 16 + (lane & 15);
                    int mb = (lane >> 4) << 3;
#pragma unroll
                    for (int k = 0; k < 8; ++k)
                        s_rad[k + mb][n] = acc[k] + s_b2[n];
                }
            }
            if (jt > 0 && t < 48) {                 // deterministic tree sum
                int h = t & 15, d = t >> 4;
                if (d < dout) {
                    float acc = 0.0f;
#pragma unroll
                    for (int jl = 0; jl < 16; ++jl) acc += s_part[jl][h][d];
                    if (dout == 1) s_msg_s[h] += acc;
                    else           s_msg_v[h][d] += acc;
                }
            }
            __syncthreads();
            // phase 2: contraction partials for this tile
            {
                int jl = t >> 4, h = t & 15;
                float p0 = 0.0f, p1 = 0.0f, p2 = 0.0f;
#pragma unroll
                for (int c = 0; c < CIN; ++c) {
                    float r = s_rad[jl][c * 16 + h];
                    p0 += r * s_G[jl][c][0];
                    p1 += r * s_G[jl][c][1];
                    p2 += r * s_G[jl][c][2];
                }
                s_part[jl][h][0] = p0; s_part[jl][h][1] = p1; s_part[jl][h][2] = p2;
            }
            __syncthreads();
        }
        // drain: tree-sum of the last tile's partials
        if (t < 48) {
            int h = t & 15, d = t >> 4;
            if (d < dout) {
                float acc = 0.0f;
#pragma unroll
                for (int jl = 0; jl < 16; ++jl) acc += s_part[jl][h][d];
                if (dout == 1) s_msg_s[h] += acc;
                else           s_msg_v[h][d] += acc;
            }
        }
        __syncthreads();
    }

    // ---- equivariant LayerNorm + write node features ----
    if (t < 16) {
        float fs = s_msg_s[t];
        s_norm[0][t] = fmaxf(fabsf(fs), EPSV);
        float vx = s_msg_v[t][0], vy = s_msg_v[t][1], vz = s_msg_v[t][2];
        s_norm[1][t] = fmaxf(sqrtf(vx * vx + vy * vy + vz * vz), EPSV);
    }
    __syncthreads();
    if (t < 2) {
        float mu = 0.0f;
        for (int h = 0; h < 16; ++h) mu += s_norm[t][h];
        mu *= (1.0f / 16.0f);
        float var = 0.0f;
        for (int h = 0; h < 16; ++h) { float d = s_norm[t][h] - mu; var += d * d; }
        var *= (1.0f / 16.0f);
        s_mu[t] = mu;
        s_isig[t] = __builtin_amdgcn_rsqf(var + 1e-5f);
    }
    __syncthreads();
    if (t < 16) {
        float ns = s_norm[0][t];
        float scl_s = (ns - s_mu[0]) * s_isig[0] * P.ln_g[0][t] + P.ln_b[0][t];
        s_out[(b * N_PTS + i) * NH + t] = s_msg_s[t] * fast_rcp(ns) * scl_s;
        float nv = s_norm[1][t];
        float scl_v = (nv - s_mu[1]) * s_isig[1] * P.ln_g[1][t] + P.ln_b[1][t];
        float f = scl_v * fast_rcp(nv);
        float* vo = v_out + ((size_t)(b * N_PTS + i) * NH + t) * 3;
        vo[0] = s_msg_v[t][0] * f;
        vo[1] = s_msg_v[t][1] * f;
        vo[2] = s_msg_v[t][2] * f;
    }
}

// ---------------------------------------------------------------------------
// Max-pool over nodes + classifier head. grid = B, 256 threads.
// ---------------------------------------------------------------------------
__global__ void __launch_bounds__(256)
head_kernel(const float* __restrict__ s_f, const float* __restrict__ v_f,
            const float* __restrict__ W0, const float* __restrict__ B0,
            const float* __restrict__ W1, const float* __restrict__ B1,
            const float* __restrict__ W2, const float* __restrict__ B2,
            float* __restrict__ out) {
    __shared__ float s_inv[N_PTS][32];
    __shared__ float s_red[8][32];
    __shared__ float z0[32], z1[64], z2[32];
    int t = threadIdx.x, b = blockIdx.x;
    {
        const float* sf = s_f + (size_t)(b * N_PTS + t) * NH;
        const float* vf = v_f + (size_t)(b * N_PTS + t) * NH * 3;
#pragma unroll
        for (int h = 0; h < 16; ++h) {
            s_inv[t][h] = sf[h];
            float x = vf[h * 3], y = vf[h * 3 + 1], z = vf[h * 3 + 2];
            s_inv[t][16 + h] = sqrtf(x * x + y * y + z * z);
        }
    }
    __syncthreads();
    {
        int k = t & 31, g = t >> 5;
        float m = -3.4e38f;
        for (int i = g * 32; i < g * 32 + 32; ++i) m = fmaxf(m, s_inv[i][k]);
        s_red[g][k] = m;
    }
    __syncthreads();
    if (t < 32) {
        float m = s_red[0][t];
#pragma unroll
        for (int g = 1; g < 8; ++g) m = fmaxf(m, s_red[g][t]);
        z0[t] = m;
    }
    __syncthreads();
    if (t < 64) {
        float a = B0[t];
        for (int k = 0; k < 32; ++k) a += z0[k] * W0[k * 64 + t];
        z1[t] = fmaxf(a, 0.0f);
    }
    __syncthreads();
    if (t < 32) {
        float a = B1[t];
        for (int k = 0; k < 64; ++k) a += z1[k] * W1[k * 32 + t];
        z2[t] = fmaxf(a, 0.0f);
    }
    __syncthreads();
    if (t < 10) {
        float a = B2[t];
        for (int k = 0; k < 32; ++k) a += z2[k] * W2[k * 10 + t];
        out[b * 10 + t] = a;
    }
}

// ---------------------------------------------------------------------------
// Launch. Input index map assumes JAX pytree flattening (dict keys sorted
// alphabetically at every level):
//   per layer L (24 leaves, base = L*24):
//     +0 ln.s.b  +1 ln.s.g  +2 ln.v.b  +3 ln.v.g
//     then per sig in {sss,svv,vsv,vvs,vvv}: W1, W2, b1, b2
//   then rho: W,b x3 (indices 72..77); then pos (78).
// ---------------------------------------------------------------------------
extern "C" void kernel_launch(void* const* d_in, const int* in_sizes, int n_in,
                              void* d_out, int out_size, void* d_ws, size_t ws_size,
                              hipStream_t stream) {
    (void)in_sizes; (void)n_in; (void)out_size; (void)ws_size;
    auto F = [&](int idx) { return (const float*)d_in[idx]; };
    const float* pos = F(3 * 24 + 6);   // index 78

    char* w = (char*)d_ws;
    auto alloc = [&](size_t bytes) {
        char* p = w; w += (bytes + 255) & ~(size_t)255; return p;
    };
    _Float16* rbf = (_Float16*)alloc((size_t)BATCH * N_PTS * N_PTS * KPAD * 2);
    float* rhat = (float*)alloc((size_t)BATCH * N_PTS * N_PTS * 3 * 4);
    float* s0 = (float*)alloc((size_t)BATCH * N_PTS * 4);
    float* v0 = (float*)alloc((size_t)BATCH * N_PTS * 3 * 4);
    float* sA = (float*)alloc((size_t)BATCH * N_PTS * NH * 4);
    float* vA = (float*)alloc((size_t)BATCH * N_PTS * NH * 3 * 4);
    float* sB = (float*)alloc((size_t)BATCH * N_PTS * NH * 4);
    float* vB = (float*)alloc((size_t)BATCH * N_PTS * NH * 3 * 4);

    geom_kernel<<<BATCH * N_PTS, 256, 0, stream>>>(pos, rbf, rhat, s0, v0);

    dim3 grid(N_PTS, BATCH);
    for (int L = 0; L < 3; ++L) {
        LayerParams P;
        int base = L * 24;
        P.ln_b[0] = F(base + 0); P.ln_g[0] = F(base + 1);
        P.ln_b[1] = F(base + 2); P.ln_g[1] = F(base + 3);
        for (int s = 0; s < 5; ++s) {
            P.W1[s] = F(base + 4 + 4 * s);
            P.W2[s] = F(base + 5 + 4 * s);
            P.b1[s] = F(base + 6 + 4 * s);
            P.b2[s] = F(base + 7 + 4 * s);
        }
        if (L == 0) {
            layer_kernel<1><<<grid, 256, 0, stream>>>(rbf, rhat, s0, v0, sA, vA, P);
        } else if (L == 1) {
            layer_kernel<16><<<grid, 256, 0, stream>>>(rbf, rhat, sA, vA, sB, vB, P);
        } else {
            layer_kernel<16><<<grid, 256, 0, stream>>>(rbf, rhat, sB, vB, sA, vA, P);
        }
    }
    int rb = 3 * 24;
    head_kernel<<<BATCH, 256, 0, stream>>>(sA, vA,
                                           F(rb + 0), F(rb + 1),
                                           F(rb + 2), F(rb + 3),
                                           F(rb + 4), F(rb + 5),
                                           (float*)d_out);
}
